// GNNPotentials_16269336117270
// MI455X (gfx1250) — matrix operand order repaired and checked by reading, not compiled, not run on hardware
//
#include <hip/hip_runtime.h>
#include <hip/hip_bf16.h>
#include <math.h>

typedef __attribute__((ext_vector_type(16))) _Float16 v16h;
typedef __attribute__((ext_vector_type(8)))  float    v8f;

#define DFEAT   16
#define NRBF_K  16
#define CUTOFF  5.0f
#define BOX_L   45.0f
#define GAMMA_W 10.0f

// ---------------------------------------------------------------------------
// Kernel 1: build transposed f16 feature table hT[d][j] = embed[z[j]][d],
// zero-padded to Npad (so garbage never reaches the WMMA B operand),
// and zero the scalar output accumulator.
// ---------------------------------------------------------------------------
__global__ void gnn_prep_kernel(const int* __restrict__ z,
                                const float* __restrict__ embed,
                                _Float16* __restrict__ hT,
                                float* __restrict__ out,
                                int N, int Npad) {
    int idx = blockIdx.x * blockDim.x + threadIdx.x;
    if (idx == 0) out[0] = 0.0f;
    int total = DFEAT * Npad;
    if (idx >= total) return;
    int d = idx / Npad;
    int j = idx - d * Npad;
    float v = 0.0f;
    if (j < N) v = embed[z[j] * DFEAT + d];
    hT[idx] = (_Float16)v;   // hT laid out [d][j], row stride Npad
}

// ---------------------------------------------------------------------------
// Kernel 2 (heavy): per-wave atom i; accumulate S_i = RBF_i(16xN) * H(Nx16)
// with v_wmma_f32_16x16x32_f16 over 32-wide j chunks, then contract with
// W_rbf to produce m[i, 0..15].
//
// A-matrix (16x32 f16) layout as built here: lane L -> row M = L&15,
//   lanes 0-15 hold K in {0..15}, lanes 16-31 hold K in {16..31};
//   v16h element e -> K = 16*half + e (same contiguous order as B).
// B-matrix (32x16 f16): lane L -> col N = L&15, K = 16*half + e  (contiguous
//   in the transposed hT row -> single 32B load per lane).
// C/D (16x16 f32): c[v] -> row M(=k) = v + 8*half, col N(=d) = L&15.
// ---------------------------------------------------------------------------
__global__ void __launch_bounds__(256)
gnn_edge_wmma_kernel(const float* __restrict__ xyz,
                     const _Float16* __restrict__ hT,
                     const float* __restrict__ W_rbf,
                     float* __restrict__ m_out,
                     int N, int Npad) {
    __shared__ float r_buf[8][32];           // one 32-float slot per wave

    const int lane = threadIdx.x & 31;
    const int wv   = threadIdx.x >> 5;
    const int i    = blockIdx.x * 8 + wv;    // one atom per wave
    if (i >= N) return;                      // wave-uniform exit

    const float xi = xyz[3 * i + 0];
    const float yi = xyz[3 * i + 1];
    const float zi = xyz[3 * i + 2];

    const int   half = lane >> 4;
    const int   d    = lane & 15;            // column role (B / C / W_rbf)
    const int   k    = lane & 15;            // RBF row role (A)
    const float mu   = (float)k * (CUTOFF / (float)(NRBF_K - 1));

    v8f c = {};                              // S_i accumulator (16x16 f32)
    const int nch = (N + 31) >> 5;

    for (int ch = 0; ch < nch; ++ch) {
        const int jbase = ch << 5;
        const int j     = jbase + lane;
        const int jl    = (j < N) ? j : (N - 1);   // clamp: no OOB, no branch

        float dx = xyz[3 * jl + 0] - xi;
        float dy = xyz[3 * jl + 1] - yi;
        float dz = xyz[3 * jl + 2] - zi;
        // minimum-image wrap (matches reference ge/lt offset logic)
        dx += (dx < -0.5f * BOX_L ? BOX_L : 0.0f) - (dx >= 0.5f * BOX_L ? BOX_L : 0.0f);
        dy += (dy < -0.5f * BOX_L ? BOX_L : 0.0f) - (dy >= 0.5f * BOX_L ? BOX_L : 0.0f);
        dz += (dz < -0.5f * BOX_L ? BOX_L : 0.0f) - (dz >= 0.5f * BOX_L ? BOX_L : 0.0f);

        const float dsq = dx * dx + dy * dy + dz * dz;
        const bool valid = (dsq < CUTOFF * CUTOFF) && (dsq > 0.0f) && (j < N);
        // sentinel r -> exp(-10*(1e4-mu)^2) underflows to exactly 0.0f,
        // reproducing the reference mask without divergence.
        const float r = valid ? sqrtf(dsq) : 1.0e4f;

        // Wave-uniform skip: ~90%+ of chunks have zero in-cutoff pairs and
        // contribute exactly 0; EXEC stays all-ones inside (WMMA requirement).
        if (__any(valid)) {
            r_buf[wv][lane] = r;             // wave-internal LDS exchange

            v16h a;
            #pragma unroll
            for (int e = 0; e < 8; ++e) {
                const float r0 = r_buf[wv][16 * half + e];
                const float r1 = r_buf[wv][16 * half + 8 + e];
                const float t0 = r0 - mu;
                const float t1 = r1 - mu;
                a[e]     = (_Float16)__expf(-GAMMA_W * t0 * t0);
                a[e + 8] = (_Float16)__expf(-GAMMA_W * t1 * t1);
            }

            // B: 16 contiguous f16 from transposed feature table (32B aligned)
            const v16h b = *(const v16h*)(hT + (size_t)d * Npad + jbase + 16 * half);

            c = __builtin_amdgcn_wmma_f32_16x16x32_f16(
                    false, a, false, b, (short)0, c, false, false);
        }
    }

    // m[i,d] = sum_k W_rbf[k,d] * S[i,k,d]; k = v + 8*half per C layout
    float partial = 0.0f;
    #pragma unroll
    for (int v = 0; v < 8; ++v)
        partial += c[v] * W_rbf[(v + 8 * half) * DFEAT + d];
    partial += __shfl_xor(partial, 16, 32);      // fold the two lane halves
    if (half == 0) m_out[i * DFEAT + d] = partial;
}

// ---------------------------------------------------------------------------
// Kernel 3 (tail): per-atom MLP + global energy reduction. ~6 MFLOP total.
// h = embed[z]; h += tanh(m @ W_msg); e = tanh(h @ W1) @ W2; out += e
// ---------------------------------------------------------------------------
__global__ void gnn_tail_kernel(const int* __restrict__ z,
                                const float* __restrict__ embed,
                                const float* __restrict__ m_in,
                                const float* __restrict__ W_msg,
                                const float* __restrict__ W1,
                                const float* __restrict__ W2,
                                float* __restrict__ out,
                                int N) {
    int i = blockIdx.x * blockDim.x + threadIdx.x;
    if (i >= N) return;

    float m[DFEAT], hn[DFEAT];
    const float* e = embed + z[i] * DFEAT;
    #pragma unroll
    for (int dd = 0; dd < DFEAT; ++dd) m[dd] = m_in[i * DFEAT + dd];

    #pragma unroll
    for (int dd = 0; dd < DFEAT; ++dd) {
        float acc = 0.0f;
        #pragma unroll
        for (int p = 0; p < DFEAT; ++p) acc += m[p] * W_msg[p * DFEAT + dd];
        hn[dd] = e[dd] + tanhf(acc);
    }

    float energy = 0.0f;
    #pragma unroll
    for (int dd = 0; dd < DFEAT; ++dd) {
        float acc = 0.0f;
        #pragma unroll
        for (int p = 0; p < DFEAT; ++p) acc += hn[p] * W1[p * DFEAT + dd];
        energy += tanhf(acc) * W2[dd];
    }
    atomicAdd(out, energy);
}

// ---------------------------------------------------------------------------
extern "C" void kernel_launch(void* const* d_in, const int* in_sizes, int n_in,
                              void* d_out, int out_size, void* d_ws, size_t ws_size,
                              hipStream_t stream) {
    const float* xyz   = (const float*)d_in[0];
    const int*   z     = (const int*)  d_in[1];
    const float* embed = (const float*)d_in[2];
    const float* W_rbf = (const float*)d_in[3];
    const float* W_msg = (const float*)d_in[4];
    const float* W1    = (const float*)d_in[5];
    const float* W2    = (const float*)d_in[6];
    float* out = (float*)d_out;

    const int N    = in_sizes[1];            // atom count (z has N entries)
    const int Npad = (N + 31) & ~31;

    // workspace layout: [hT f16 16*Npad][align 256][m f32 N*16]
    _Float16* hT = (_Float16*)d_ws;
    size_t hT_bytes = (size_t)DFEAT * Npad * sizeof(_Float16);
    size_t m_off    = (hT_bytes + 255) & ~(size_t)255;
    float* m_buf    = (float*)((char*)d_ws + m_off);

    {   // prep: hT + zero output
        int total = DFEAT * Npad;
        int blocks = (total + 255) / 256;
        gnn_prep_kernel<<<blocks, 256, 0, stream>>>(z, embed, hT, out, N, Npad);
    }
    {   // heavy pairwise WMMA: 8 waves (8 atoms) per 256-thread block
        int blocks = (N + 7) / 8;
        gnn_edge_wmma_kernel<<<blocks, 256, 0, stream>>>(xyz, hT, W_rbf, m_buf, N, Npad);
    }
    {   // tail MLP + reduction
        int blocks = (N + 127) / 128;
        gnn_tail_kernel<<<blocks, 128, 0, stream>>>(z, embed, m_buf, W_msg, W1, W2, out, N);
    }
}
